// MoE_35777077576177
// MI455X (gfx1250) — matrix operand order repaired and checked by reading, not compiled, not run on hardware
//
#include <hip/hip_runtime.h>
#include <hip/hip_bf16.h>

typedef __attribute__((ext_vector_type(16))) _Float16 v16h;
typedef __attribute__((ext_vector_type(8)))  float    v8f;

#define WMMA_F16(Aop, Bop, Cacc) \
    __builtin_amdgcn_wmma_f32_16x16x32_f16(false, (Aop), false, (Bop), (short)0, (Cacc), false, false)

#define B_TOTAL 524288
#define N_CHUNK (B_TOTAL / 32)   // 32 rows (2 x 16-col tiles) per wave-iteration

// LDS operand table: ops 0..63 expert W1, ops 64..65 gating W1, op 66 gating W2
#define N_OPS   67
#define OP_GW1  64
#define OP_GW2  66

static __device__ __forceinline__ unsigned pkh2(float a, float b) {
    _Float16 ha = (_Float16)a, hb = (_Float16)b;
    unsigned short ua, ub;
    __builtin_memcpy(&ua, &ha, 2);
    __builtin_memcpy(&ub, &hb, 2);
    return (unsigned)ua | ((unsigned)ub << 16);
}

union U16H { uint4 q[2]; v16h v; };

__global__ __launch_bounds__(256) void moe_kernel(
    const float* __restrict__ A_data, const float* __restrict__ S_data,
    const float* __restrict__ gw1, const float* __restrict__ gb1,
    const float* __restrict__ gw2, const float* __restrict__ gb2,
    const float* __restrict__ ew1, const float* __restrict__ eb1,
    const float* __restrict__ ew2, const float* __restrict__ eb2,
    float* __restrict__ out)
{
    // f16 WMMA A-operand layout, one 1KB block per operand:
    // uint4 index = op*64 + q*32 + lane  (q = v>>2), dword component = v&3
    __shared__ uint4 sW1[N_OPS * 64];    // 67 KB
    __shared__ uint4 sG[512];            // 8 KB: per-wave gating relayout scratch

    const int tid  = threadIdx.x;
    const int lane = tid & 31;
    const int wave = tid >> 5;
    const int lh   = lane >> 4;          // lane half (0: lanes 0-15, 1: 16-31)
    const int lm   = lane & 15;

    // ---- one-time: all weights -> LDS in f16 A-operand layout ----
    unsigned* sW1d = (unsigned*)sW1;
    #pragma unroll 1
    for (int d = tid; d < N_OPS * 256; d += 256) {
        int op  = d >> 8;                // 256 dwords per operand
        int q   = (d >> 7) & 1;
        int ln  = (d >> 2) & 31;
        int vc  = d & 3;
        int v   = q * 4 + vc;
        int lh2 = ln >> 4;
        int lmm = ln & 15;
        // A layout: K = 16*(v>>2) + 2*(v&3) + s + 8*(lane>=16)
        int kb  = 16 * (v >> 2) + 2 * (v & 3) + 8 * lh2;
        float f0, f1;
        if (op < 64) {                           // expert W1^T tiles
            int e = op >> 3, m = (op >> 1) & 3, ks = op & 1;
            int h  = m * 16 + lmm;
            int k0 = ks * 32 + kb;
            f0 = ew1[(e * 64 + k0    ) * 64 + h];
            f1 = ew1[(e * 64 + k0 + 1) * 64 + h];
        } else if (op < 66) {                    // gating W1^T tiles (K=32)
            int m = op - 64;
            int h = m * 16 + lmm;
            f0 = gw1[(kb    ) * 32 + h];
            f1 = gw1[(kb + 1) * 32 + h];
        } else {                                 // gating W2^T (E rows, zero-padded)
            if (lmm < 8) { f0 = gw2[kb * 8 + lmm]; f1 = gw2[(kb + 1) * 8 + lmm]; }
            else         { f0 = 0.0f;             f1 = 0.0f; }
        }
        sW1d[d] = pkh2(f0, f1);
    }

    // ---- per-lane invariant biases (small, kept in VGPRs) ----
    v8f gb1l[2], gb2l;
    #pragma unroll
    for (int m = 0; m < 2; ++m)
        #pragma unroll
        for (int r = 0; r < 8; ++r)
            gb1l[m][r] = gb1[m * 16 + 8 * lh + r];
    #pragma unroll
    for (int r = 0; r < 8; ++r) gb2l[r] = gb2[r];
    float eb2l[8];
    #pragma unroll
    for (int e = 0; e < 8; ++e) eb2l[e] = eb2[e];

    __syncthreads();

    const int stride = gridDim.x * 8;
    #pragma unroll 1
    for (int chunk = blockIdx.x * 8 + wave; chunk < N_CHUNK; chunk += stride) {
        const int r0 = chunk * 32;

        // prefetch next chunk's rows
        int nxt = chunk + stride;
        if (nxt < N_CHUNK) {
            __builtin_prefetch((const char*)(A_data + (nxt * 32 + lane) * 32), 0, 3);
            __builtin_prefetch((const char*)(S_data + (nxt * 32 + lane) * 32), 0, 3);
        }

        // ---- B operands: X^T tiles, f32 -> f16. B layout: K = 16*(v>>2)+2*(v&3)+s+8*lh, N = lm ----
        v16h Bx[2][2];
        #pragma unroll
        for (int t = 0; t < 2; ++t)
            #pragma unroll
            for (int ks = 0; ks < 2; ++ks) {
                const float* src = (ks == 0 ? A_data : S_data) + (r0 + t * 16 + lm) * 32 + 8 * lh;
                const float4* s4 = (const float4*)src;
                float4 q0 = s4[0], q1 = s4[1], q2 = s4[4], q3 = s4[5];
                float fv[16] = { q0.x,q0.y,q0.z,q0.w, q1.x,q1.y,q1.z,q1.w,
                                 q2.x,q2.y,q2.z,q2.w, q3.x,q3.y,q3.z,q3.w };
                v16h b;
                #pragma unroll
                for (int i = 0; i < 16; ++i) b[i] = (_Float16)fv[i];
                Bx[t][ks] = b;
            }

        // ---- gating: weight operands from LDS (shared), 3 WMMA per tile ----
        float pr[2][8];
        {
            U16H ga0, ga1, ga2;
            ga0.q[0] = sW1[OP_GW1 * 64 + lane];       ga0.q[1] = sW1[OP_GW1 * 64 + 32 + lane];
            ga1.q[0] = sW1[(OP_GW1 + 1) * 64 + lane]; ga1.q[1] = sW1[(OP_GW1 + 1) * 64 + 32 + lane];
            ga2.q[0] = sW1[OP_GW2 * 64 + lane];       ga2.q[1] = sW1[OP_GW2 * 64 + 32 + lane];
            const int wbase = wave * 64;
            #pragma unroll
            for (int t = 0; t < 2; ++t) {
                v8f g0 = gb1l[0], g1 = gb1l[1];
                g0 = WMMA_F16(ga0.v, Bx[t][1], g0);
                g1 = WMMA_F16(ga1.v, Bx[t][1], g1);
                #pragma unroll
                for (int r = 0; r < 8; ++r) { g0[r] = fmaxf(g0[r], 0.0f); g1[r] = fmaxf(g1[r], 0.0f); }
                uint4 u0, u1;
                u0.x = pkh2(g0[0], g0[1]); u0.y = pkh2(g0[2], g0[3]);
                u0.z = pkh2(g0[4], g0[5]); u0.w = pkh2(g0[6], g0[7]);
                u1.x = pkh2(g1[0], g1[1]); u1.y = pkh2(g1[2], g1[3]);
                u1.z = pkh2(g1[4], g1[5]); u1.w = pkh2(g1[6], g1[7]);
                // C layout (h in vgpr r + 8*lh + 16*m, b in lane) -> B-operand rows
                sG[wbase + lh * 16 + lm]      = u0;   // m=0 -> own lane, q[0]
                sG[wbase + 32 + lh * 16 + lm] = u1;   // m=1 -> own lane, q[1]
                asm volatile("s_wait_dscnt 0" ::: "memory");
                U16H gB;
                gB.q[0] = sG[wbase + lane];
                gB.q[1] = sG[wbase + 32 + lane];
                v8f lg = gb2l;
                lg = WMMA_F16(ga2.v, gB.v, lg);       // logits: E in vgpr r (lanes 0-15)
                float mx = lg[0];
                #pragma unroll
                for (int r = 1; r < 8; ++r) mx = fmaxf(mx, lg[r]);
                float sum = 0.0f, p[8];
                #pragma unroll
                for (int r = 0; r < 8; ++r) { p[r] = __expf(lg[r] - mx); sum += p[r]; }
                float inv = 1.0f / sum;
                #pragma unroll
                for (int r = 0; r < 8; ++r) pr[t][r] = p[r] * inv;
            }
        }

        // ---- experts: EH^T = W1e^T @ X^T (K=64), relu, dot w2, weighted sum ----
        // e fully unrolled (pr/eb2l stay const-indexed); m rolled to bound pressure.
        float outv0 = 0.0f, outv1 = 0.0f;
        #pragma unroll
        for (int e = 0; e < 8; ++e) {
            float p0 = 0.0f, p1 = 0.0f;
            #pragma unroll 1
            for (int m = 0; m < 4; ++m) {
                const float4* bp = (const float4*)(eb1 + e * 64 + m * 16 + 8 * lh);
                float4 ba = bp[0], bb = bp[1];
                float bv[8] = { ba.x, ba.y, ba.z, ba.w, bb.x, bb.y, bb.z, bb.w };
                v8f c0, c1;
                #pragma unroll
                for (int r = 0; r < 8; ++r) { c0[r] = bv[r]; c1[r] = bv[r]; }
                const int opb = (e * 4 + m) * 2;
                U16H a0, a1;
                a0.q[0] = sW1[opb * 64 + lane];       a0.q[1] = sW1[opb * 64 + 32 + lane];
                a1.q[0] = sW1[(opb + 1) * 64 + lane]; a1.q[1] = sW1[(opb + 1) * 64 + 32 + lane];
                c0 = WMMA_F16(a0.v, Bx[0][0], c0);
                c0 = WMMA_F16(a1.v, Bx[0][1], c0);
                c1 = WMMA_F16(a0.v, Bx[1][0], c1);
                c1 = WMMA_F16(a1.v, Bx[1][1], c1);
                const float4* wp = (const float4*)(ew2 + e * 64 + m * 16 + 8 * lh);
                float4 wa = wp[0], wb = wp[1];
                float wv[8] = { wa.x, wa.y, wa.z, wa.w, wb.x, wb.y, wb.z, wb.w };
                #pragma unroll
                for (int r = 0; r < 8; ++r) {
                    p0 += fmaxf(c0[r], 0.0f) * wv[r];
                    p1 += fmaxf(c1[r], 0.0f) * wv[r];
                }
            }
            // combine the two hidden half-sets (lane l <-> l^16), add output bias
            float f0 = p0 + __shfl_xor(p0, 16) + eb2l[e];
            float f1 = p1 + __shfl_xor(p1, 16) + eb2l[e];
            outv0 += pr[0][e] * f0;
            outv1 += pr[1][e] * f1;
        }

        if (lane < 16) {
            out[r0 + lane]      = outv0;
            out[r0 + 16 + lane] = outv1;
        }
    }
}

extern "C" void kernel_launch(void* const* d_in, const int* in_sizes, int n_in,
                              void* d_out, int out_size, void* d_ws, size_t ws_size,
                              hipStream_t stream) {
    (void)in_sizes; (void)n_in; (void)out_size; (void)d_ws; (void)ws_size;
    const float* A_data = (const float*)d_in[0];
    const float* S_data = (const float*)d_in[1];
    const float* gw1    = (const float*)d_in[2];
    const float* gb1    = (const float*)d_in[3];
    const float* gw2    = (const float*)d_in[4];
    const float* gb2    = (const float*)d_in[5];
    const float* ew1    = (const float*)d_in[6];
    const float* eb1    = (const float*)d_in[7];
    const float* ew2    = (const float*)d_in[8];
    const float* eb2    = (const float*)d_in[9];
    float* outp = (float*)d_out;

    moe_kernel<<<512, 256, 0, stream>>>(A_data, S_data, gw1, gb1, gw2, gb2,
                                        ew1, eb1, ew2, eb2, outp);
}
